// BiLSTMCRF_78529182040555
// MI455X (gfx1250) — matrix operand order repaired and checked by reading, not compiled, not run on hardware
//
#include <hip/hip_runtime.h>
#include <hip/hip_bf16.h>

typedef _Float16 v16h __attribute__((ext_vector_type(16)));
typedef _Float16 v8h  __attribute__((ext_vector_type(8)));
typedef float    v8f  __attribute__((ext_vector_type(8)));

#define Vv   20000
#define Ee   100
#define Hh   200
#define Kk   17
#define Bb   128
#define Tt   512

#define EP   128           // E padded to 4 K-chunks of 32
#define HP   224           // H padded to 7 K-chunks of 32 (also per-gate column pad)
#define NP   896           // 4 * HP gate columns
#define NB   56            // NP / 16 column blocks
#define KC_T 11            // 4 (x) + 7 (h) K-chunks of 32
#define RES_KC    5        // K-chunks resident in LDS (all of x + first h chunk)
#define CHUNK_ELEMS (NB * 32 * 16)                   // f16 per K-chunk = 28672
#define WCAT_PER_DIR ((size_t)KC_T * CHUNK_ELEMS)    // f16 elements per direction

__device__ __forceinline__ float sigmoidf_(float x) { return 1.0f / (1.0f + expf(-x)); }

__device__ __forceinline__ v16h make_frag(v8h lo, v8h hi) {
    union { v16h v; v8h h[2]; } u;
    u.h[0] = lo; u.h[1] = hi;
    return u.v;
}

// ---------------------------------------------------------------- prep kernels

__global__ void cvt_emb_kernel(const float* __restrict__ emb, _Float16* __restrict__ dst) {
    int idx = blockIdx.x * blockDim.x + threadIdx.x;
    if (idx >= Vv * EP) return;
    int v = idx >> 7, e = idx & 127;
    dst[idx] = (e < Ee) ? (_Float16)emb[v * Ee + e] : (_Float16)0.0f;
}

// Pack W (row-major [4H, Ksrc]) into WMMA B-fragment layout:
// fragment (kc,nb): 32 lanes x 16 f16; lane holds column n = nb*16 + (lane&15),
// K values kbase..kbase+15 with kbase = kc*32 + (lane<16 ? 0 : 16).
__global__ void pack_w_kernel(const float* __restrict__ W, int Ksrc, int kcCount, int kcOff,
                              _Float16* __restrict__ dst) {
    int idx = blockIdx.x * blockDim.x + threadIdx.x;
    int total = kcCount * NB * 32;
    if (idx >= total) return;
    int lane = idx & 31;
    int nb   = (idx >> 5) % NB;
    int kcl  = idx / (NB * 32);
    int n  = nb * 16 + (lane & 15);
    int q  = n / HP, jn = n % HP;
    int n_orig = q * Hh + jn;
    int kbase  = kcl * 32 + ((lane < 16) ? 0 : 16);
    _Float16* out = dst + ((((size_t)(kcl + kcOff) * NB + nb) * 32 + lane) << 4);
#pragma unroll
    for (int j = 0; j < 16; ++j) {
        int k = kbase + j;
        float v = (jn < Hh && k < Ksrc) ? W[(size_t)n_orig * Ksrc + k] : 0.0f;
        out[j] = (_Float16)v;
    }
}

__global__ void pad_bias_kernel(const float* __restrict__ b, float* __restrict__ dst) {
    int idx = blockIdx.x * blockDim.x + threadIdx.x;
    if (idx >= NP) return;
    int q = idx / HP, jn = idx % HP;
    dst[idx] = (jn < Hh) ? b[q * Hh + jn] : 0.0f;
}

// ---------------------------------------------------------------- BiLSTM recurrence
// grid (8, 2): x = batch tile of 16 rows, y = direction. 224 threads = 7 waves.
// Each wave owns 2 hidden-blocks (32 hidden units) across all 4 gates.
// K-chunks 0..4 of the packed weights live in LDS; chunks 5..10 stream from L2.
__global__ __launch_bounds__(224, 1)
void bilstm_kernel(const int* __restrict__ sent, const _Float16* __restrict__ embh,
                   const _Float16* __restrict__ Wcat, const float* __restrict__ bpad,
                   float* __restrict__ lstm_out) {
    __shared__ _Float16 sWres[RES_KC * CHUNK_ELEMS];  // 286,720 B
    __shared__ _Float16 sA[16][368];     // cols 0..127: x (f16), cols 128..351: h (f16)
    __shared__ float    sHout[16][200];

    const int tid  = threadIdx.x;
    const int lane = tid & 31;
    const int wid  = tid >> 5;           // 0..6
    const int hb0  = wid * 2;
    const int col  = lane & 15;
    const int half = lane >> 4;
    const int dir  = blockIdx.y;
    const int b0   = blockIdx.x * 16;

    const _Float16* wc = Wcat + (size_t)dir * WCAT_PER_DIR;
    const float*    bp = bpad + (size_t)dir * NP;
    const _Float16* wstream = wc + (size_t)RES_KC * CHUNK_ELEMS;

    // stage resident weight chunks into LDS (uint4 = 8 f16 per copy)
    {
        const uint4* src = (const uint4*)wc;
        uint4* dstv = (uint4*)sWres;
        for (int i = tid; i < RES_KC * CHUNK_ELEMS / 8; i += 224) dstv[i] = src[i];
    }
    // zero h region of sA
    for (int i = tid; i < 16 * HP; i += 224) sA[i / HP][128 + (i % HP)] = (_Float16)0.0f;

    float creg[2][8] = {};
    float bqv[4][2];
#pragma unroll
    for (int q = 0; q < 4; ++q)
#pragma unroll
        for (int hh = 0; hh < 2; ++hh)
            bqv[q][hh] = bp[q * HP + (hb0 + hh) * 16 + col];

    for (int step = 0; step < Tt; ++step) {
        const int t = (dir == 0) ? step : (Tt - 1 - step);

        // opaque copy: prevents hoisting of the streamed weight loads out of the
        // step loop (which previously caused a full spill of B fragments to scratch)
        const _Float16* ws_ = wstream;
        asm volatile("" : "+s"(ws_));

        // gather x rows (f16, padded to 128) into sA[:,0:128]
        for (int i = tid; i < 16 * 32; i += 224) {
            int m = i >> 5, seg = i & 31;
            int tok = sent[(b0 + m) * Tt + t];
            *(uint2*)&sA[m][seg * 4] = *(const uint2*)(embh + (size_t)tok * EP + seg * 4);
        }
        __syncthreads();   // x ready; h from previous step ready

        v8f acc[8] = {};   // index q*2 + hh

        // ---- LDS-resident K-chunks (x chunks 0..3 + h chunk 4)
        for (int kc = 0; kc < RES_KC; ++kc) {
            int k0 = kc * 32 + ((lane < 16) ? 0 : 16);
            const _Float16* ar = &sA[col][0];
            v16h afrag = make_frag(*(const v8h*)(ar + k0), *(const v8h*)(ar + k0 + 8));
            // batch all 8 B-fragment loads before the 8 WMMAs so the scheduler can
            // use staged s_wait_dscnt values instead of draining per-WMMA
            v16h bfr[8];
#pragma unroll
            for (int q = 0; q < 4; ++q)
#pragma unroll
                for (int hh = 0; hh < 2; ++hh) {
                    int nb = q * 14 + hb0 + hh;
                    bfr[q * 2 + hh] = *(const v16h*)(sWres + (((kc * NB + nb) * 32 + lane) << 4));
                }
#pragma unroll
            for (int f = 0; f < 8; ++f)
                acc[f] = __builtin_amdgcn_wmma_f32_16x16x32_f16(
                    false, afrag, false, bfr[f], (short)0, acc[f], false, false);
        }
        // ---- L2-streamed K-chunks (h chunks 5..10)
        for (int kc = RES_KC; kc < KC_T; ++kc) {
            int k0 = kc * 32 + ((lane < 16) ? 0 : 16);
            const _Float16* ar = &sA[col][0];
            v16h afrag = make_frag(*(const v8h*)(ar + k0), *(const v8h*)(ar + k0 + 8));
            v16h bfr[8];
#pragma unroll
            for (int q = 0; q < 4; ++q)
#pragma unroll
                for (int hh = 0; hh < 2; ++hh) {
                    int nb = q * 14 + hb0 + hh;
                    bfr[q * 2 + hh] = *(const v16h*)(ws_ + ((((size_t)(kc - RES_KC) * NB + nb) * 32 + lane) << 4));
                }
#pragma unroll
            for (int f = 0; f < 8; ++f)
                acc[f] = __builtin_amdgcn_wmma_f32_16x16x32_f16(
                    false, afrag, false, bfr[f], (short)0, acc[f], false, false);
        }

        // LSTM cell (torch gate order i, f, g, o)
        float hreg[2][8];
#pragma unroll
        for (int hh = 0; hh < 2; ++hh) {
#pragma unroll
            for (int r = 0; r < 8; ++r) {
                float gi = acc[0 * 2 + hh][r] + bqv[0][hh];
                float gf = acc[1 * 2 + hh][r] + bqv[1][hh];
                float gg = acc[2 * 2 + hh][r] + bqv[2][hh];
                float go = acc[3 * 2 + hh][r] + bqv[3][hh];
                float c  = creg[hh][r];
                c = sigmoidf_(gf) * c + sigmoidf_(gi) * tanhf(gg);
                float h = sigmoidf_(go) * tanhf(c);
                creg[hh][r] = c;
                hreg[hh][r] = h;
            }
        }
        __syncthreads();   // all waves done reading sA

#pragma unroll
        for (int hh = 0; hh < 2; ++hh) {
#pragma unroll
            for (int r = 0; r < 8; ++r) {
                int M = r + half * 8;
                int n = (hb0 + hh) * 16 + col;
                sA[M][128 + n] = (_Float16)hreg[hh][r];
                if (n < Hh) sHout[M][n] = hreg[hh][r];
            }
        }
        __syncthreads();   // new h staged

        // coalesced store of h tile to lstm_out[t][b0+m][dir*H + n]
        for (int i = tid; i < 16 * Hh; i += 224) {
            int m = i / Hh, n = i % Hh;
            lstm_out[((size_t)t * Bb + (b0 + m)) * (2 * Hh) + dir * Hh + n] = sHout[m][n];
        }
    }
}

// ---------------------------------------------------------------- emissions
// One wave per (t,b) row; lanes 0..16 each compute one tag's dot over 2H=400.
__global__ __launch_bounds__(256)
void emissions_kernel(const float* __restrict__ lstm_out, const float* __restrict__ W_fc,
                      const float* __restrict__ b_fc, float* __restrict__ em) {
    __shared__ float sW[Kk][401];
    __shared__ float sB[Kk];
    int tid = threadIdx.x;
    for (int i = tid; i < Kk * 400; i += 256) sW[i / 400][i % 400] = W_fc[i];
    if (tid < Kk) sB[tid] = b_fc[tid];
    __syncthreads();

    int wid  = tid >> 5, lane = tid & 31;
    int row  = blockIdx.x * 8 + wid;                // 0..65535
    int kk   = (lane < Kk) ? lane : (Kk - 1);
    const float* x = lstm_out + (size_t)row * 400;
    float s = sB[kk];
#pragma unroll 4
    for (int j = 0; j < 400; ++j) s += x[j] * sW[kk][j];
    if (lane < Kk) em[(size_t)row * Kk + lane] = s;
}

// ---------------------------------------------------------------- bin head
__global__ __launch_bounds__(256)
void bin_kernel(const float* __restrict__ lstm_out, const float* __restrict__ W_bin,
                const float* __restrict__ b_bin, float* __restrict__ out) {
    __shared__ float ssum[400];
    int b = blockIdx.x, tid = threadIdx.x;
    for (int j = tid; j < 400; j += 256) {
        float acc = 0.0f;
        for (int t = 0; t < Tt; ++t) acc += lstm_out[((size_t)t * Bb + b) * 400 + j];
        ssum[j] = acc * (1.0f / (float)Tt);
    }
    __syncthreads();
    if (tid < 2) {
        float s = b_bin[tid];
        for (int j = 0; j < 400; ++j) s += ssum[j] * W_bin[tid * 400 + j];
        out[(size_t)Bb * Tt + b * 2 + tid] = s;
    }
}

// ---------------------------------------------------------------- Viterbi
// One wave per batch element; lanes 0..16 hold per-tag scores.
__global__ __launch_bounds__(128)
void viterbi_kernel(const float* __restrict__ em, const int* __restrict__ masks,
                    const float* __restrict__ trans, const float* __restrict__ start,
                    const float* __restrict__ end, int* __restrict__ bpbuf,
                    float* __restrict__ out_tags) {
    __shared__ float sTT[Kk][Kk + 1];   // sTT[k][prev] = trans[prev][k]
    int tid = threadIdx.x;
    for (int i = tid; i < Kk * Kk; i += 128) {
        int k = i % Kk, p = i / Kk;
        sTT[k][p] = trans[p * Kk + k];
    }
    __syncthreads();

    int lane = tid & 31, wid = tid >> 5;
    int b = blockIdx.x * 4 + wid;
    int kk = (lane < Kk) ? lane : (Kk - 1);

    float score = start[kk] + em[(size_t)(0 * Bb + b) * Kk + kk];
    if (lane >= Kk) score = -3.0e38f;

    for (int t = 1; t < Tt; ++t) {
        int m = masks[b * Tt + t];
        float emv = em[((size_t)t * Bb + b) * Kk + kk];
        float best = -3.0e38f;
        int bi = kk;
#pragma unroll
        for (int prev = 0; prev < Kk; ++prev) {
            float sp = __shfl(score, prev);
            float cand = sp + sTT[kk][prev];
            if (cand > best) { best = cand; bi = prev; }
        }
        int bpv;
        if (m != 0) { score = best + emv; bpv = bi; }
        else        { bpv = kk; }
        if (lane < Kk) bpbuf[((size_t)(t - 1) * Bb + b) * Kk + lane] = bpv;
    }

    // argmax over lanes of score + end (first max wins)
    float v = (lane < Kk) ? (score + end[kk]) : -3.0e38f;
    int bi = lane;
#pragma unroll
    for (int off = 16; off > 0; off >>= 1) {
        float ov = __shfl_xor(v, off);
        int   oi = __shfl_xor(bi, off);
        if (ov > v || (ov == v && oi < bi)) { v = ov; bi = oi; }
    }

    if (lane == 0) {
        int cur = bi;
        out_tags[(size_t)b * Tt + (Tt - 1)] = (float)cur;
        for (int tt = Tt - 2; tt >= 0; --tt) {
            cur = bpbuf[((size_t)tt * Bb + b) * Kk + cur];
            out_tags[(size_t)b * Tt + tt] = (float)cur;
        }
    }
}

// ---------------------------------------------------------------- launcher

static inline size_t align256(size_t x) { return (x + 255) & ~(size_t)255; }

extern "C" void kernel_launch(void* const* d_in, const int* in_sizes, int n_in,
                              void* d_out, int out_size, void* d_ws, size_t ws_size,
                              hipStream_t stream) {
    const int*   sentence = (const int*)d_in[0];
    const int*   masks    = (const int*)d_in[1];
    const float* emb      = (const float*)d_in[2];
    const float* W_ih_f   = (const float*)d_in[3];
    const float* W_hh_f   = (const float*)d_in[4];
    const float* b_f      = (const float*)d_in[5];
    const float* W_ih_b   = (const float*)d_in[6];
    const float* W_hh_b   = (const float*)d_in[7];
    const float* b_b      = (const float*)d_in[8];
    const float* W_fc     = (const float*)d_in[9];
    const float* b_fc     = (const float*)d_in[10];
    const float* W_bin    = (const float*)d_in[11];
    const float* b_bin    = (const float*)d_in[12];
    const float* trans    = (const float*)d_in[13];
    const float* start    = (const float*)d_in[14];
    const float* endw     = (const float*)d_in[15];

    char* ws = (char*)d_ws;
    size_t off = 0;
    _Float16* embh = (_Float16*)(ws + off); off = align256(off + (size_t)Vv * EP * sizeof(_Float16));
    _Float16* Wcat = (_Float16*)(ws + off); off = align256(off + 2 * WCAT_PER_DIR * sizeof(_Float16));
    float* bpad    = (float*)(ws + off);    off = align256(off + 2 * NP * sizeof(float));
    float* lstm    = (float*)(ws + off);    off = align256(off + (size_t)Tt * Bb * 2 * Hh * sizeof(float));
    float* em      = (float*)(ws + off);    off = align256(off + (size_t)Tt * Bb * Kk * sizeof(float));
    int*   bpbuf   = (int*)(ws + off);      off = align256(off + (size_t)(Tt - 1) * Bb * Kk * sizeof(int));

    float* out = (float*)d_out;

    // 1. f16 embedding table (padded E -> 128)
    cvt_emb_kernel<<<(Vv * EP + 255) / 256, 256, 0, stream>>>(emb, embh);

    // 2. pack weights into WMMA B-fragment layout (x chunks kc 0..3, h chunks kc 4..10)
    int nIH = 4 * NB * 32, nHH = 7 * NB * 32;
    pack_w_kernel<<<(nIH + 255) / 256, 256, 0, stream>>>(W_ih_f, Ee, 4, 0, Wcat);
    pack_w_kernel<<<(nHH + 255) / 256, 256, 0, stream>>>(W_hh_f, Hh, 7, 4, Wcat);
    pack_w_kernel<<<(nIH + 255) / 256, 256, 0, stream>>>(W_ih_b, Ee, 4, 0, Wcat + WCAT_PER_DIR);
    pack_w_kernel<<<(nHH + 255) / 256, 256, 0, stream>>>(W_hh_b, Hh, 7, 4, Wcat + WCAT_PER_DIR);
    pad_bias_kernel<<<(NP + 255) / 256, 256, 0, stream>>>(b_f, bpad);
    pad_bias_kernel<<<(NP + 255) / 256, 256, 0, stream>>>(b_b, bpad + NP);

    // 3. persistent BiLSTM recurrence (WMMA f16->f32, LDS-resident + L2-streamed weights)
    bilstm_kernel<<<dim3(Bb / 16, 2), 224, 0, stream>>>(sentence, embh, Wcat, bpad, lstm);

    // 4. emissions + bin head
    emissions_kernel<<<(Tt * Bb) / 8, 256, 0, stream>>>(lstm, W_fc, b_fc, em);
    bin_kernel<<<Bb, 256, 0, stream>>>(lstm, W_bin, b_bin, out);

    // 5. Viterbi decode
    viterbi_kernel<<<Bb / 4, 128, 0, stream>>>(em, masks, trans, start, endw, bpbuf, out);
}
